// ElasticityTGN_LE_76046690943362
// MI455X (gfx1250) — compile-verified, matrix operand order checked by reading, not run on hardware
//
#include <hip/hip_runtime.h>
#include <hip/hip_bf16.h>
#include <math.h>

#define N_NODES 16384
#define E_BASE  131072
#define K_EXTRA 5                      /* K_EXTRA+1 in python terms */
#define E_EXTRA (N_NODES * K_EXTRA)    /* 81920 */
#define E_TOT   (E_BASE + E_EXTRA)     /* 212992 */
#define D_LAT   64

typedef __attribute__((ext_vector_type(16))) __bf16 v16bf;
typedef __attribute__((ext_vector_type(8)))  float  v8f;

// ---------------------------------------------------------------------------
// Generic fused WMMA GEMM:  OUT[M x N] = epilogue( A[M x K] @ W[Kact x N] + b )
// 256 threads = 8 waves; block computes a 64-row x 64-col macro-tile.
// REQUIRES: M % 64 == 0 and N % 64 == 0 (true for every launch here), so the
// epilogue and staging are completely guard-free straight-line code.
// Wave w: n-subtile = w&3, m-subtiles = (w>>2)*2 + {0,1}  (2 accumulators).
// A assembled on the fly from up to 3 f32 sources (optional row-gather),
// bf16-converted into LDS. W (f32) is bf16-converted into LDS in *fragment
// order* so each B fragment is one contiguous 32B LDS read (2x ds_load_b128).
// KPAD and MODE are compile-time: K-loop fully unrolled (2*KPAD/32 WMMAs per
// wave).
// Modes: 0 plain, 1 relu, 2 relu+add_row, 3 relu(add_mat + relu(.)),
//        4 atomic scatter-add of relu(.)*scale[row] into out[sidx[row]*N+n]
// ---------------------------------------------------------------------------
template<int KPAD, int MODE>
__global__ __launch_bounds__(256)
void wmma_gemm_kernel(const float* __restrict__ Wt, int Kact, int N,
                      const float* __restrict__ bias,
                      const float* __restrict__ s0, int c0, const int* __restrict__ i0,
                      const float* __restrict__ s1, int c1, const int* __restrict__ i1,
                      const float* __restrict__ s2, int c2, const int* __restrict__ i2,
                      float* __restrict__ out,
                      const float* __restrict__ add_row,
                      const float* __restrict__ add_mat,
                      const float* __restrict__ scale,
                      const int* __restrict__ sidx)
{
    constexpr int KTILES = KPAD / 32;
    __shared__ __align__(32) __bf16 sA[64 * KPAD];              // row-major, stride KPAD
    __shared__ __align__(32) __bf16 sB[KTILES * 4 * 32 * 16];   // fragment-order

    const int m0   = blockIdx.x * 64;
    const int nc   = blockIdx.y * 64;
    const int tid  = threadIdx.x;
    const int lane = tid & 31;
    const int wv   = tid >> 5;                // 0..7
    const int ntile = wv & 3;                 // n-subtile 0..3
    const int ms0   = ((wv >> 2) << 1) * 16;  // 0 or 32: first m-subtile row offset

    // ---- stage A panel (64 x KPAD): 4 threads per row, shift/mask math only ----
    {
        const int m   = tid >> 2;
        const int row = m0 + m;
        const int cs01  = c0 + c1;
        const int cs012 = cs01 + c2;
        for (int k = (tid & 3); k < KPAD; k += 4) {
            float v = 0.f;
            if (k < c0)         v = s0[(i0 ? i0[row] : row) * c0 + k];
            else if (k < cs01)  v = s1[(i1 ? i1[row] : row) * c1 + (k - c0)];
            else if (k < cs012) v = s2[(i2 ? i2[row] : row) * c2 + (k - cs01)];
            sA[m * KPAD + k] = (__bf16)v;
        }
    }
    // ---- stage B panel in fragment order: idx = ((kb*4+t)*32+ln)*16+e ----
    for (int idx = tid; idx < KPAD * 64; idx += 256) {
        int e   = idx & 15;
        int ln  = (idx >> 4) & 31;
        int t   = (idx >> 9) & 3;
        int kb  = idx >> 11;
        int k   = kb * 32 + e + (e >= 8 ? 8 : 0) + ((ln >> 4) << 3);
        int n   = nc + t * 16 + (ln & 15);
        float v = (k < Kact) ? Wt[k * N + n] : 0.f;
        sB[idx] = (__bf16)v;
    }
    __syncthreads();

    v8f acc0 = {}, acc1 = {};
    const int mrow = lane & 15;
    const int kg   = (lane >> 4) << 3;
#pragma unroll
    for (int kb = 0; kb < KTILES; ++kb) {
        v16bf b = *(const v16bf*)&sB[((kb * 4 + ntile) * 32 + lane) * 16];
        v16bf a0, a1;
        const int kbase = kb * 32 + kg;
#pragma unroll
        for (int e = 0; e < 16; ++e) {
            int k = kbase + e + (e >= 8 ? 8 : 0);
            a0[e] = sA[(ms0 + mrow) * KPAD + k];
            a1[e] = sA[(ms0 + 16 + mrow) * KPAD + k];
        }
        acc0 = __builtin_amdgcn_wmma_f32_16x16x32_bf16(
                   false, a0, false, b, (short)0, acc0, false, false);
        acc1 = __builtin_amdgcn_wmma_f32_16x16x32_bf16(
                   false, a1, false, b, (short)0, acc1, false, false);
    }

    // ---- guard-free epilogue for both m-subtiles ----
    const int nn = nc + ntile * 16 + (lane & 15);
    const float bval = bias[nn];
    const float arow = (MODE == 2) ? add_row[nn] : 0.f;
#pragma unroll
    for (int j = 0; j < 2; ++j) {
        const v8f& acc = (j == 0) ? acc0 : acc1;
        const int mbase = m0 + ms0 + j * 16 + ((lane >> 4) << 3);
#pragma unroll
        for (int r = 0; r < 8; ++r) {
            const int row = mbase + r;
            float v = acc[r] + bval;
            if (MODE == 0) {
                out[row * N + nn] = v;
            } else if (MODE == 1) {
                out[row * N + nn] = fmaxf(v, 0.f);
            } else if (MODE == 2) {
                out[row * N + nn] = fmaxf(v, 0.f) + arow;
            } else if (MODE == 3) {
                out[row * N + nn] = fmaxf(add_mat[row * N + nn] + fmaxf(v, 0.f), 0.f);
            } else { // 4: fused masked segment-sum scatter
                atomicAdd(&out[sidx[row] * N + nn], fmaxf(v, 0.f) * scale[row]);
            }
        }
    }
}

// ---- tiny global-feature encoder: g = lin1(relu(lin0(gf))) -----------------
__global__ __launch_bounds__(64)
void glob_kernel(const float* gf, const float* w0, const float* b0,
                 const float* w1, const float* b1, float* g)
{
    __shared__ float h[64];
    int j = threadIdx.x;
    float v = gf[0]*w0[j] + gf[1]*w0[64+j] + gf[2]*w0[128+j] + b0[j];
    h[j] = fmaxf(v, 0.f);
    __syncthreads();
    float acc = b1[j];
    for (int k = 0; k < 64; ++k) acc += h[k] * w1[k * 64 + j];
    g[j] = acc;
}

// ---- brute-force KNN top-5 over 2D coords (LDS-tiled) ----------------------
__global__ __launch_bounds__(256)
void knn_kernel(const float* __restrict__ x, int* __restrict__ knn_t)
{
    __shared__ float2 tile[256];
    int i = blockIdx.x * 256 + threadIdx.x;
    float xi = x[i * 3 + 0], yi = x[i * 3 + 1];
    float bd[5]; int bj[5];
#pragma unroll
    for (int s = 0; s < 5; ++s) { bd[s] = 3.4e38f; bj[s] = i; }
    for (int t0 = 0; t0 < N_NODES; t0 += 256) {
        int j = t0 + threadIdx.x;
        tile[threadIdx.x] = make_float2(x[j * 3 + 0], x[j * 3 + 1]);
        __syncthreads();
        for (int jj = 0; jj < 256; ++jj) {
            float dx = tile[jj].x - xi, dy = tile[jj].y - yi;
            float d = dx * dx + dy * dy;
            if (d < bd[4]) {
                int jid = t0 + jj, p = 4;
                while (p > 0 && d < bd[p - 1]) {
                    bd[p] = bd[p - 1]; bj[p] = bj[p - 1]; --p;
                }
                bd[p] = d; bj[p] = jid;
            }
        }
        __syncthreads();
    }
#pragma unroll
    for (int s = 0; s < 5; ++s) knn_t[i * 5 + s] = bj[s];
}

__global__ void copy_base_kernel(const int* ei, const float* ef,
                                 int* srcA, int* tgtA, float* emask, float* efeat)
{
    int e = blockIdx.x * blockDim.x + threadIdx.x;
    if (e >= E_BASE) return;
    srcA[e] = ei[e];
    tgtA[e] = ei[E_BASE + e];
    emask[e] = 1.f;
    efeat[e * 2 + 0] = ef[e * 2 + 0];
    efeat[e * 2 + 1] = ef[e * 2 + 1];
}

__global__ void build_extra_kernel(const float* x, const int* knn_t,
                                   int* srcA, int* tgtA, float* efeat)
{
    int e = blockIdx.x * blockDim.x + threadIdx.x;
    if (e >= E_EXTRA) return;
    int s = e / 5, t = knn_t[e];
    srcA[E_BASE + e] = s;
    tgtA[E_BASE + e] = t;
    efeat[(E_BASE + e) * 2 + 0] = x[t * 3 + 0] - x[s * 3 + 0];
    efeat[(E_BASE + e) * 2 + 1] = x[t * 3 + 1] - x[s * 3 + 1];
}

// ---- infer layer2 (64->1) + sigmoid gate -> emask --------------------------
__global__ void infer_gate_kernel(const float* __restrict__ h,
                                  const float* w1, const float* b1,
                                  const int* srcA, const int* tgtA, float* emask)
{
    int e = blockIdx.x * blockDim.x + threadIdx.x;
    if (e >= E_EXTRA) return;
    float acc = b1[0];
    for (int k = 0; k < 64; ++k) acc += h[e * 64 + k] * w1[k];
    float w = 1.f / (1.f + expf(-acc));
    bool keep = (tgtA[E_BASE + e] != srcA[E_BASE + e]) && (w > 0.5f);
    emask[E_BASE + e] = keep ? 1.f : 0.f;
}

__global__ void zero_kernel(float* p, int n)
{
    int i = blockIdx.x * blockDim.x + threadIdx.x;
    if (i < n) p[i] = 0.f;
}

// ---- GRU pointwise with mem0 == 0 (gh reduces to b_hh) ---------------------
__global__ void gru_pointwise_kernel(const float* __restrict__ gi,
                                     const float* __restrict__ bhh,
                                     float* __restrict__ mem)
{
    int idx = blockIdx.x * blockDim.x + threadIdx.x;
    if (idx >= N_NODES * 64) return;
    int i = idx >> 6, d = idx & 63;
    float r = 1.f / (1.f + expf(-(gi[i * 192 + d]        + bhh[d])));
    float z = 1.f / (1.f + expf(-(gi[i * 192 + 64 + d]   + bhh[64 + d])));
    float n = tanhf(gi[i * 192 + 128 + d] + r * bhh[128 + d]);
    mem[idx] = (1.f - z) * n;
}

__global__ void dec_tail_kernel(const float* __restrict__ h,
                                const float* w, const float* b, float* out)
{
    int idx = blockIdx.x * blockDim.x + threadIdx.x;
    if (idx >= N_NODES * 3) return;
    int i = idx / 3, o = idx - i * 3;
    float acc = b[o];
    for (int k = 0; k < 64; ++k) acc += h[i * 64 + k] * w[k * 3 + o];
    out[idx] = acc;
}

// ---------------------------------------------------------------------------
template<int KPAD, int MODE>
static inline void gemm_launch(hipStream_t st, dim3 g,
                               const float* Wt, int Kact, int N, const float* bias,
                               const float* s0, int c0, const int* i0,
                               const float* s1, int c1, const int* i1,
                               const float* s2, int c2, const int* i2,
                               float* out,
                               const float* add_row, const float* add_mat,
                               const float* scale, const int* sidx)
{
    wmma_gemm_kernel<KPAD, MODE><<<g, 256, 0, st>>>(Wt, Kact, N, bias,
        s0, c0, i0, s1, c1, i1, s2, c2, i2, out, add_row, add_mat, scale, sidx);
}

static inline void gemm(hipStream_t st, const float* Wt, int Kact, int Kpad, int N,
                        const float* bias,
                        const float* s0, int c0, const int* i0,
                        const float* s1, int c1, const int* i1,
                        const float* s2, int c2, const int* i2,
                        int M, float* out,
                        const float* add_row, const float* add_mat,
                        const float* scale, const int* sidx, int mode)
{
    dim3 g(M / 64, N / 64);   // M, N always multiples of 64 in this pipeline
#define GL(KP, MD) gemm_launch<KP, MD>(st, g, Wt, Kact, N, bias, s0, c0, i0, \
        s1, c1, i1, s2, c2, i2, out, add_row, add_mat, scale, sidx)
    if (Kpad == 32) {
        GL(32, 1);
    } else if (Kpad == 64) {
        switch (mode) {
        case 0: GL(64, 0); break;
        case 1: GL(64, 1); break;
        case 2: GL(64, 2); break;
        case 3: GL(64, 3); break;
        default: GL(64, 4); break;
        }
    } else if (Kpad == 128) {
        GL(128, 1);
    } else {
        GL(160, 1);
    }
#undef GL
}

extern "C" void kernel_launch(void* const* d_in, const int* in_sizes, int n_in,
                              void* d_out, int out_size, void* d_ws, size_t ws_size,
                              hipStream_t stream)
{
    (void)in_sizes; (void)n_in; (void)out_size; (void)ws_size;
    const float* x  = (const float*)d_in[0];
    const int*   ei = (const int*)d_in[1];
    const float* ef = (const float*)d_in[2];
    const float* gf = (const float*)d_in[3];
    // params leaves, pytree-sorted keys: dec, edge_enc, glob_enc, gru, infer,
    // msg, node_enc, upd; each linear flattens as (b, w).
    auto P = [&](int i) { return (const float*)d_in[4 + i]; };
    const float *dec0_b=P(0),  *dec0_w=P(1),  *dec1_b=P(2),  *dec1_w=P(3),
                *dec2_b=P(4),  *dec2_w=P(5);
    const float *ee0_b=P(6),   *ee0_w=P(7),   *ee1_b=P(8),   *ee1_w=P(9);
    const float *ge0_b=P(10),  *ge0_w=P(11),  *ge1_b=P(12),  *ge1_w=P(13);
    const float *gru_bhh=P(14),*gru_bih=P(15),*gru_wih=P(17); /* w_hh unused (mem0=0) */
    const float *inf0_b=P(18), *inf0_w=P(19), *inf1_b=P(20), *inf1_w=P(21);
    const float *msg_b0[3]={P(22),P(26),P(30)}, *msg_w0[3]={P(23),P(27),P(31)};
    const float *msg_b1[3]={P(24),P(28),P(32)}, *msg_w1[3]={P(25),P(29),P(33)};
    const float *ne0_b=P(34),  *ne0_w=P(35),  *ne1_b=P(36),  *ne1_w=P(37);
    const float *upd_b0[3]={P(38),P(42),P(46)}, *upd_w0[3]={P(39),P(43),P(47)};
    const float *upd_b1[3]={P(40),P(44),P(48)}, *upd_w1[3]={P(41),P(45),P(49)};

    // workspace carve-up (f32 units)
    float* W = (float*)d_ws;
    size_t o = 0;
    float* g_vec = W + o; o += 64;
    float* x_enc = W + o; o += (size_t)N_NODES * 64;
    float* h1    = W + o; o += (size_t)N_NODES * 64;   // node hidden / dec hidden1
    int*   knn_t = (int*)(W + o); o += E_EXTRA;
    int*   srcA  = (int*)(W + o); o += E_TOT;
    int*   tgtA  = (int*)(W + o); o += E_TOT;
    float* emask = W + o; o += E_TOT;
    float* efeat = W + o; o += (size_t)E_TOT * 2;
    float* e_enc = W + o; o += (size_t)E_TOT * 64;
    float* hbig  = W + o; o += (size_t)E_TOT * 64;     // infer/edge/msg hidden
    float* m_v   = W + o; o += (size_t)N_NODES * 64;
    float* h_upd = W + o; o += (size_t)N_NODES * 64;
    float* gi    = W + o; o += (size_t)N_NODES * 192;
    float* hd2   = W + o; o += (size_t)N_NODES * 64;

    float* out_final = (float*)d_out;                   // 16384 x 3
    float* mem       = out_final + (size_t)N_NODES * 3; // 16384 x 64

    // 1. global encoder
    glob_kernel<<<1, 64, 0, stream>>>(gf, ge0_w, ge0_b, ge1_w, ge1_b, g_vec);
    // 2. node encoder: relu(x@W0+b0) -> relu(.@W1+b1) + g
    gemm(stream, ne0_w, 3, 32, 64, ne0_b, x, 3, nullptr,
         nullptr, 0, nullptr, nullptr, 0, nullptr,
         N_NODES, h1, nullptr, nullptr, nullptr, nullptr, 1);
    gemm(stream, ne1_w, 64, 64, 64, ne1_b, h1, 64, nullptr,
         nullptr, 0, nullptr, nullptr, 0, nullptr,
         N_NODES, x_enc, g_vec, nullptr, nullptr, nullptr, 2);
    // 3. KNN + edge assembly
    knn_kernel<<<N_NODES / 256, 256, 0, stream>>>(x, knn_t);
    copy_base_kernel<<<(E_BASE + 255) / 256, 256, 0, stream>>>(ei, ef, srcA, tgtA, emask, efeat);
    build_extra_kernel<<<(E_EXTRA + 255) / 256, 256, 0, stream>>>(x, knn_t, srcA, tgtA, efeat);
    // 4. edge-inference MLP (K = 64+64+2 = 130, padded to 160) + sigmoid gate
    gemm(stream, inf0_w, 130, 160, 64, inf0_b,
         x_enc, 64, srcA + E_BASE,
         x_enc, 64, tgtA + E_BASE,
         efeat + (size_t)E_BASE * 2, 2, nullptr,
         E_EXTRA, hbig, nullptr, nullptr, nullptr, nullptr, 1);
    infer_gate_kernel<<<(E_EXTRA + 255) / 256, 256, 0, stream>>>(hbig, inf1_w, inf1_b, srcA, tgtA, emask);
    // 5. edge encoder over all 212992 edges
    gemm(stream, ee0_w, 2, 32, 64, ee0_b, efeat, 2, nullptr,
         nullptr, 0, nullptr, nullptr, 0, nullptr,
         E_TOT, hbig, nullptr, nullptr, nullptr, nullptr, 1);
    gemm(stream, ee1_w, 64, 64, 64, ee1_b, hbig, 64, nullptr,
         nullptr, 0, nullptr, nullptr, 0, nullptr,
         E_TOT, e_enc, nullptr, nullptr, nullptr, nullptr, 1);
    // 6. three message-passing rounds (layer2 fused with masked segment-sum)
    for (int rd = 0; rd < 3; ++rd) {
        zero_kernel<<<(N_NODES * 64 + 255) / 256, 256, 0, stream>>>(m_v, N_NODES * 64);
        gemm(stream, msg_w0[rd], 128, 128, 64, msg_b0[rd],
             x_enc, 64, srcA, e_enc, 64, nullptr, nullptr, 0, nullptr,
             E_TOT, hbig, nullptr, nullptr, nullptr, nullptr, 1);
        gemm(stream, msg_w1[rd], 64, 64, 64, msg_b1[rd],
             hbig, 64, nullptr, nullptr, 0, nullptr, nullptr, 0, nullptr,
             E_TOT, m_v, nullptr, nullptr, emask, tgtA, 4);
        gemm(stream, upd_w0[rd], 128, 128, 64, upd_b0[rd],
             x_enc, 64, nullptr, m_v, 64, nullptr, nullptr, 0, nullptr,
             N_NODES, h_upd, nullptr, nullptr, nullptr, nullptr, 1);
        gemm(stream, upd_w1[rd], 64, 64, 64, upd_b1[rd],
             h_upd, 64, nullptr, nullptr, 0, nullptr, nullptr, 0, nullptr,
             N_NODES, x_enc, nullptr, x_enc, nullptr, nullptr, 3);
    }
    // 7. GRU: gi = x_enc @ w_ih + b_ih (N=192), then pointwise with b_hh
    gemm(stream, gru_wih, 64, 64, 192, gru_bih,
         x_enc, 64, nullptr, nullptr, 0, nullptr, nullptr, 0, nullptr,
         N_NODES, gi, nullptr, nullptr, nullptr, nullptr, 0);
    gru_pointwise_kernel<<<(N_NODES * 64 + 255) / 256, 256, 0, stream>>>(gi, gru_bhh, mem);
    // 8. decoder
    gemm(stream, dec0_w, 64, 64, 64, dec0_b, mem, 64, nullptr,
         nullptr, 0, nullptr, nullptr, 0, nullptr,
         N_NODES, h1, nullptr, nullptr, nullptr, nullptr, 1);
    gemm(stream, dec1_w, 64, 64, 64, dec1_b, h1, 64, nullptr,
         nullptr, 0, nullptr, nullptr, 0, nullptr,
         N_NODES, hd2, nullptr, nullptr, nullptr, nullptr, 1);
    dec_tail_kernel<<<(N_NODES * 3 + 255) / 256, 256, 0, stream>>>(hd2, dec2_w, dec2_b, out_final);
}
